// SuperPoint_66812511257203
// MI455X (gfx1250) — compile-verified, matrix operand order checked by reading, not run on hardware
//
#include <hip/hip_runtime.h>
#include <hip/hip_bf16.h>
#include <stdint.h>

typedef __attribute__((ext_vector_type(16))) _Float16 v16h;
typedef __attribute__((ext_vector_type(8)))  _Float16 v8h;
typedef __attribute__((ext_vector_type(8)))  float    v8f;

__device__ __forceinline__ v16h cat16(v8h lo, v8h hi) {
    return __builtin_shufflevector(lo, hi, 0,1,2,3,4,5,6,7,8,9,10,11,12,13,14,15);
}

// ---------------------------------------------------------------------------
// Utility: f32 -> f16 convert ; zero-fill helper
// ---------------------------------------------------------------------------
__global__ void cvt_f32_f16(const float* __restrict__ in, _Float16* __restrict__ out, int n) {
    int i = blockIdx.x * blockDim.x + threadIdx.x;
    if (i < n) out[i] = (_Float16)in[i];
}
__global__ void zero_f16(_Float16* __restrict__ p, int n) {
    int i = blockIdx.x * blockDim.x + threadIdx.x;
    if (i < n) p[i] = (_Float16)0.f;
}

// ---------------------------------------------------------------------------
// Pack conv weights [Cout,Cin,KH,KW] (f32) into [CoutPad, Kpad] f16 rows,
// TAP-MAJOR K order: k' = tap*Cin + cin  (tap = ky*KW + kx).
// ---------------------------------------------------------------------------
__global__ void pack_weights(const float* __restrict__ w, _Float16* __restrict__ wpk,
                             int Cout, int Cin, int Kreal, int Kpad, int CoutPad) {
    int i = blockIdx.x * blockDim.x + threadIdx.x;
    int total = CoutPad * Kpad;
    if (i >= total) return;
    int co = i / Kpad, k = i - co * Kpad;
    float v = 0.f;
    if (co < Cout && k < Kreal) {
        int khw = Kreal / Cin;
        int tap = k / Cin;
        int cin = k - tap * Cin;
        v = w[(size_t)co * Kreal + (size_t)cin * khw + tap];
    }
    wpk[i] = (_Float16)v;
}

// ---------------------------------------------------------------------------
// WMMA fragments (ISA 7.12.2, wave32, 16-bit A 16x32):
//   A lane L: M=L&15; elems 0..7 -> K = kb+(L>=16?8:0)+e ; 8..15 -> +16.
//   B mirrors with lane = N (pixel). C/D: VGPR r: M=r+8*(L>=16), N=L&15.
// Activations are NHWC so a B fragment = two contiguous 16B loads.
// ---------------------------------------------------------------------------

// 3x3 conv, Cin = 1<<CINL2, NHWC activations, tap-major packed weights.
template <int NT, int CINL2>
__global__ void __launch_bounds__(256)
conv3x3_wmma(const _Float16* __restrict__ x, const _Float16* __restrict__ wpk,
             const float* __restrict__ bias, _Float16* __restrict__ out16,
             float* __restrict__ out32, const _Float16* __restrict__ zerobuf,
             int Bn, int Cout, int CoutPad, int H, int W,
             int do_relu, int f32out) {
    constexpr int Cin  = 1 << CINL2;
    constexpr int Kpad = 9 * Cin;

    const int lane = threadIdx.x & 31;
    const int wave = (int)((blockIdx.x * blockDim.x + threadIdx.x) >> 5);
    const int HW = H * W;
    const int mTiles = CoutPad >> 4;
    const int nTiles = HW / (16 * NT);
    const int tilesPerB = mTiles * nTiles;
    if (wave >= Bn * tilesPerB) return;

    const int b  = wave / tilesPerB;
    const int t  = wave - b * tilesPerB;
    const int mt = t / nTiles;
    const int nt = t - mt * nTiles;

    const int mLo = mt << 4;
    const int hi  = lane >> 4;
    const int kLaneOff = hi << 3;

    int pp[NT], pyA[NT], pxA[NT];
#pragma unroll
    for (int j = 0; j < NT; ++j) {
        int p  = nt * (16 * NT) + j * 16 + (lane & 15);
        int py = p / W;
        pp[j] = p; pyA[j] = py; pxA[j] = p - py * W;
    }

    const _Float16* wrow = wpk + (size_t)(mLo + (lane & 15)) * Kpad;
    const _Float16* xb   = x + (size_t)b * HW * Cin;     // NHWC

    v8f acc[NT];
#pragma unroll
    for (int j = 0; j < NT; ++j) acc[j] = (v8f){0.f,0.f,0.f,0.f,0.f,0.f,0.f,0.f};

#pragma unroll
    for (int ky = 0; ky < 3; ++ky) {
#pragma unroll
        for (int kx = 0; kx < 3; ++kx) {
            const _Float16* wtap = wrow + (ky * 3 + kx) * Cin;
            const _Float16* bptr[NT];
#pragma unroll
            for (int j = 0; j < NT; ++j) {
                int iy = pyA[j] + ky - 1, ix = pxA[j] + kx - 1;
                bool ok = ((unsigned)iy < (unsigned)H) && ((unsigned)ix < (unsigned)W);
                long long boff = (long long)(iy * W + ix) * Cin * (int)sizeof(_Float16);
                uintptr_t pa = (uintptr_t)xb + (intptr_t)boff;
                bptr[j] = ok ? (const _Float16*)pa : zerobuf;
            }
#pragma unroll
            for (int kb = 0; kb < Cin; kb += 32) {
                v16h a = cat16(*(const v8h*)(wtap + kb + kLaneOff),
                               *(const v8h*)(wtap + kb + 16 + kLaneOff));
#pragma unroll
                for (int j = 0; j < NT; ++j) {
                    v16h bm = cat16(*(const v8h*)(bptr[j] + kb + kLaneOff),
                                    *(const v8h*)(bptr[j] + kb + 16 + kLaneOff));
                    acc[j] = __builtin_amdgcn_wmma_f32_16x16x32_f16(
                        false, a, false, bm, (short)0, acc[j], false, false);
                }
            }
        }
    }

    // epilogue: NHWC stores, each lane owns 8 consecutive couts
    float bv[8];
    const int coBase = mLo + (hi << 3);
#pragma unroll
    for (int r = 0; r < 8; ++r) bv[r] = (coBase + r < Cout) ? bias[coBase + r] : 0.f;
#pragma unroll
    for (int j = 0; j < NT; ++j) {
        if (f32out) {
#pragma unroll
            for (int r = 0; r < 8; ++r) {
                if (coBase + r < Cout) {
                    float v = acc[j][r] + bv[r];
                    if (do_relu && v < 0.f) v = 0.f;
                    out32[(size_t)pp[j] * Cout + coBase + r + (size_t)b * HW * Cout] = v;
                }
            }
        } else {   // Cout multiple of 16 for all f16 layers -> vector store
            v8h st;
#pragma unroll
            for (int r = 0; r < 8; ++r) {
                float v = acc[j][r] + bv[r];
                if (do_relu && v < 0.f) v = 0.f;
                st[r] = (_Float16)v;
            }
            *(v8h*)(out16 + (size_t)b * HW * Cout + (size_t)pp[j] * Cout + coBase) = st;
        }
    }
}

// 3x3 conv with Cin==1 (first layer). img has a padded margin; taps clamped so
// loads are always in-bounds, then zero-selected (branchless).
template <int NT>
__global__ void __launch_bounds__(256)
conv3x3_c1_wmma(const _Float16* __restrict__ x, const _Float16* __restrict__ wpk,
                const float* __restrict__ bias, _Float16* __restrict__ out16,
                int Cout, int H, int W, int Bn) {
    constexpr int Kpad = 32;
    const int lane = threadIdx.x & 31;
    const int wave = (int)((blockIdx.x * blockDim.x + threadIdx.x) >> 5);
    const int HW = H * W;
    const int nTiles = HW / (16 * NT);
    const int tilesPerB = (Cout >> 4) * nTiles;
    if (wave >= Bn * tilesPerB) return;

    const int b  = wave / tilesPerB;
    const int t  = wave - b * tilesPerB;
    const int mt = t / nTiles;
    const int nt = t - mt * nTiles;
    const int mLo = mt << 4;
    const int hi  = lane >> 4;
    const int kLaneOff = hi << 3;

    const _Float16* wrow = wpk + (size_t)(mLo + (lane & 15)) * Kpad;
    const _Float16* xb   = x + (size_t)b * HW;
    v16h a = cat16(*(const v8h*)(wrow + kLaneOff), *(const v8h*)(wrow + 16 + kLaneOff));

    const int coBase = mLo + (hi << 3);
#pragma unroll
    for (int j = 0; j < NT; ++j) {
        int p  = nt * (16 * NT) + j * 16 + (lane & 15);
        int py = p / W, px = p - (p / W) * W;
        v16h bm;
#pragma unroll
        for (int e = 0; e < 16; ++e) {
            int k  = kLaneOff + ((e < 8) ? e : e + 8);
            int kc = (k < 9) ? k : 8;                // clamp -> safe address
            int ky = kc / 3, kx = kc - ky * 3;
            int iy = py + ky - 1, ix = px + kx - 1;
            _Float16 v = xb[(long long)iy * W + ix]; // margin-padded buffer
            bool ok = (k < 9) && ((unsigned)iy < (unsigned)H) && ((unsigned)ix < (unsigned)W);
            bm[e] = ok ? v : (_Float16)0.f;
        }
        v8f c = (v8f){0.f,0.f,0.f,0.f,0.f,0.f,0.f,0.f};
        c = __builtin_amdgcn_wmma_f32_16x16x32_f16(false, a, false, bm,
                                                   (short)0, c, false, false);
        v8h st;
#pragma unroll
        for (int r = 0; r < 8; ++r) {
            float v = c[r] + bias[coBase + r];
            if (v < 0.f) v = 0.f;
            st[r] = (_Float16)v;
        }
        *(v8h*)(out16 + (size_t)b * HW * Cout + (size_t)p * Cout + coBase) = st;
    }
}

// 1x1 conv (heads), NHWC: both fragments are contiguous vector loads.
template <int NT, int CIN>
__global__ void __launch_bounds__(256)
conv1x1_wmma(const _Float16* __restrict__ x, const _Float16* __restrict__ wpk,
             const float* __restrict__ bias, _Float16* __restrict__ out16,
             float* __restrict__ out32,
             int Bn, int Cout, int CoutPad, int H, int W,
             int do_relu, int f32out) {
    const int lane = threadIdx.x & 31;
    const int wave = (int)((blockIdx.x * blockDim.x + threadIdx.x) >> 5);
    const int HW = H * W;
    const int mTiles = CoutPad >> 4;
    const int nTiles = HW / (16 * NT);
    const int tilesPerB = mTiles * nTiles;
    if (wave >= Bn * tilesPerB) return;

    const int b  = wave / tilesPerB;
    const int t  = wave - b * tilesPerB;
    const int mt = t / nTiles;
    const int nt = t - mt * nTiles;
    const int mLo = mt << 4;
    const int hi  = lane >> 4;
    const int kLaneOff = hi << 3;

    const _Float16* wrow = wpk + (size_t)(mLo + (lane & 15)) * CIN;
    const _Float16* xp[NT];
    int pp[NT];
#pragma unroll
    for (int j = 0; j < NT; ++j) {
        pp[j] = nt * (16 * NT) + j * 16 + (lane & 15);
        xp[j] = x + ((size_t)b * HW + pp[j]) * CIN;
    }

    v8f acc[NT];
#pragma unroll
    for (int j = 0; j < NT; ++j) acc[j] = (v8f){0.f,0.f,0.f,0.f,0.f,0.f,0.f,0.f};

#pragma unroll
    for (int kb = 0; kb < CIN; kb += 32) {
        v16h a = cat16(*(const v8h*)(wrow + kb + kLaneOff),
                       *(const v8h*)(wrow + kb + 16 + kLaneOff));
#pragma unroll
        for (int j = 0; j < NT; ++j) {
            v16h bm = cat16(*(const v8h*)(xp[j] + kb + kLaneOff),
                            *(const v8h*)(xp[j] + kb + 16 + kLaneOff));
            acc[j] = __builtin_amdgcn_wmma_f32_16x16x32_f16(
                false, a, false, bm, (short)0, acc[j], false, false);
        }
    }

    float bv[8];
    const int coBase = mLo + (hi << 3);
#pragma unroll
    for (int r = 0; r < 8; ++r) bv[r] = (coBase + r < Cout) ? bias[coBase + r] : 0.f;
#pragma unroll
    for (int j = 0; j < NT; ++j) {
        if (f32out) {
#pragma unroll
            for (int r = 0; r < 8; ++r) {
                if (coBase + r < Cout) {
                    float v = acc[j][r] + bv[r];
                    if (do_relu && v < 0.f) v = 0.f;
                    out32[((size_t)b * HW + pp[j]) * Cout + coBase + r] = v;
                }
            }
        } else {
            v8h st;
#pragma unroll
            for (int r = 0; r < 8; ++r) {
                float v = acc[j][r] + bv[r];
                if (do_relu && v < 0.f) v = 0.f;
                st[r] = (_Float16)v;
            }
            *(v8h*)(out16 + ((size_t)b * HW + pp[j]) * Cout + coBase) = st;
        }
    }
}

// ---------------------------------------------------------------------------
// 2x2 max pool, stride 2, f16, NHWC
// ---------------------------------------------------------------------------
__global__ void pool2_f16(const _Float16* __restrict__ in, _Float16* __restrict__ out,
                          int Bn, int C, int H, int W) {
    int Ho = H >> 1, Wo = W >> 1;
    int i = blockIdx.x * blockDim.x + threadIdx.x;
    int total = Bn * Ho * Wo * C;
    if (i >= total) return;
    int c = i % C; int t = i / C;
    int w = t % Wo; t /= Wo;
    int h = t % Ho; int b = t / Ho;
    const _Float16* p = in + (((size_t)b * H + 2 * h) * W + 2 * w) * C + c;
    float a = (float)p[0], bb = (float)p[C], cc = (float)p[(size_t)W * C], d = (float)p[(size_t)W * C + C];
    out[i] = (_Float16)fmaxf(fmaxf(a, bb), fmaxf(cc, d));
}

// ---------------------------------------------------------------------------
// Detector head: softmax over 65 contiguous channels (NHWC), drop dustbin,
// pixel-shuffle 8x8.  raw: [B,h,w,65] f32 ; sc: [B, h*8, w*8] f32
// ---------------------------------------------------------------------------
__global__ void score_head(const float* __restrict__ raw, float* __restrict__ sc,
                           int Bn, int h, int w) {
    int cells = h * w;
    int i = blockIdx.x * blockDim.x + threadIdx.x;
    if (i >= Bn * cells) return;
    int b = i / cells, cell = i - b * cells;
    int y = cell / w, x = cell - y * w;
    const float* base = raw + ((size_t)b * cells + cell) * 65;
    float m = -1e30f;
    for (int c = 0; c < 65; ++c) m = fmaxf(m, base[c]);
    float s = 0.f;
    for (int c = 0; c < 65; ++c) s += __expf(base[c] - m);
    float inv = 1.f / s;
    int H8 = h * 8, W8 = w * 8;
    for (int c = 0; c < 64; ++c) {
        float pv = __expf(base[c] - m) * inv;
        int row = y * 8 + (c >> 3);
        int col = x * 8 + (c & 7);
        sc[((size_t)b * H8 + row) * W8 + col] = pv;
    }
}

// ---------------------------------------------------------------------------
// 7x7 max pool, stride 1, SAME (-inf padding) on [B,H,W] f32
// ---------------------------------------------------------------------------
__global__ void maxpool7(const float* __restrict__ in, float* __restrict__ out,
                         int Bn, int H, int W) {
    int i = blockIdx.x * blockDim.x + threadIdx.x;
    if (i >= Bn * H * W) return;
    int x = i % W; int t = i / W; int y = t % H; int b = t / H;
    const float* pb = in + (size_t)b * H * W;
    float m = -__builtin_inff();
    for (int dy = -3; dy <= 3; ++dy) {
        int yy = y + dy; if (yy < 0 || yy >= H) continue;
        for (int dx = -3; dx <= 3; ++dx) {
            int xx = x + dx; if (xx < 0 || xx >= W) continue;
            m = fmaxf(m, pb[(size_t)yy * W + xx]);
        }
    }
    out[i] = m;
}

__global__ void nms_mask_init(const float* __restrict__ sc, const float* __restrict__ mp,
                              float* __restrict__ m, int n) {
    int i = blockIdx.x * blockDim.x + threadIdx.x;
    if (i < n) m[i] = (sc[i] == mp[i]) ? 1.f : 0.f;
}
__global__ void nms_supp(const float* __restrict__ sc, const float* __restrict__ mpm,
                         float* __restrict__ s1, int n) {
    int i = blockIdx.x * blockDim.x + threadIdx.x;
    if (i < n) s1[i] = (mpm[i] > 0.f) ? 0.f : sc[i];
}
__global__ void nms_update(float* __restrict__ m, const float* __restrict__ s1,
                           const float* __restrict__ mps1, const float* __restrict__ mpm, int n) {
    int i = blockIdx.x * blockDim.x + threadIdx.x;
    if (i >= n) return;
    if ((s1[i] == mps1[i]) && !(mpm[i] > 0.f)) m[i] = 1.f;
}
__global__ void nms_final(const float* __restrict__ sc, const float* __restrict__ m,
                          float* __restrict__ out, int n) {
    int i = blockIdx.x * blockDim.x + threadIdx.x;
    if (i < n) out[i] = (m[i] > 0.f) ? sc[i] : 0.f;
}

// ---------------------------------------------------------------------------
// Descriptor: L2-normalize 256 contiguous channels (NHWC input) and write
// transposed to NCHW d_out.  raw: [B,HW,256] ; out: [B,256,HW]
// ---------------------------------------------------------------------------
__global__ void desc_norm_t(const float* __restrict__ raw, float* __restrict__ out,
                            int Bn, int C, int HW) {
    int i = blockIdx.x * blockDim.x + threadIdx.x;
    if (i >= Bn * HW) return;
    int b = i / HW, cell = i - b * HW;
    const float* p = raw + (size_t)i * C;
    float s = 0.f;
    for (int c = 0; c < C; ++c) { float v = p[c]; s += v * v; }
    float inv = 1.f / sqrtf(s);
    float* o = out + (size_t)b * C * HW + cell;
    for (int c = 0; c < C; ++c) o[(size_t)c * HW] = p[c] * inv;
}

// ---------------------------------------------------------------------------
// Host-side orchestration
// ---------------------------------------------------------------------------
static inline int ceil_div(int a, int b) { return (a + b - 1) / b; }

extern "C" void kernel_launch(void* const* d_in, const int* in_sizes, int n_in,
                              void* d_out, int out_size, void* d_ws, size_t ws_size,
                              hipStream_t stream) {
    (void)in_sizes; (void)n_in; (void)out_size; (void)ws_size;
    const int B = 8;
    const int H0 = 480, W0 = 640;
    const int NPIX = B * H0 * W0;
    constexpr int NT = 4;

    const float* image = (const float*)d_in[0];
    float* out_sc   = (float*)d_out;
    float* out_nms  = out_sc + (size_t)NPIX;
    float* out_desc = out_nms + (size_t)NPIX;

    uint8_t* ws = (uint8_t*)d_ws;
    size_t off = 0;
    auto alloc = [&](size_t bytes) -> void* {
        void* p = ws + off;
        off += (bytes + 255) & ~(size_t)255;
        return p;
    };

    const size_t MARGIN = 8192;   // bytes; covers +-(W+1) f16 for conv1a taps
    _Float16* img16 = (_Float16*)((uint8_t*)alloc((size_t)NPIX * 2 + 2 * MARGIN) + MARGIN);
    const size_t ACT_ELEMS = (size_t)B * 64 * H0 * W0;
    _Float16* act0 = (_Float16*)alloc(ACT_ELEMS * 2);
    _Float16* act1 = (_Float16*)alloc(ACT_ELEMS * 2);
    _Float16* zerobuf = (_Float16*)alloc(1024);

    struct WP { int Cin, Cout, KH; _Float16* pk; int Kreal, Kpad, CoutPad; };
    WP wp[12] = {
        {  1,  64, 3}, { 64,  64, 3},
        { 64,  64, 3}, { 64,  64, 3},
        { 64, 128, 3}, {128, 128, 3},
        {128, 128, 3}, {128, 128, 3},
        {128, 256, 3}, {256,  65, 1},
        {128, 256, 3}, {256, 256, 1},
    };
    for (int l = 0; l < 12; ++l) {
        wp[l].Kreal   = wp[l].Cin * wp[l].KH * wp[l].KH;
        wp[l].Kpad    = (wp[l].Kreal + 31) & ~31;
        wp[l].CoutPad = (wp[l].Cout + 15) & ~15;
        wp[l].pk = (_Float16*)alloc((size_t)wp[l].CoutPad * wp[l].Kpad * 2);
    }

    float* scraw   = (float*)alloc((size_t)B * 60 * 80 * 65 * 4);   // NHWC
    float* descRaw = (float*)alloc((size_t)B * 60 * 80 * 256 * 4);  // NHWC
    float* Mb = (float*)alloc((size_t)NPIX * 4);
    float* T0 = (float*)alloc((size_t)NPIX * 4);
    float* T1 = (float*)alloc((size_t)NPIX * 4);
    float* S1 = (float*)alloc((size_t)NPIX * 4);

    // stage 0
    zero_f16<<<2, 256, 0, stream>>>(zerobuf, 512);
    cvt_f32_f16<<<ceil_div(NPIX, 256), 256, 0, stream>>>(image, img16, NPIX);
    for (int l = 0; l < 12; ++l) {
        int total = wp[l].CoutPad * wp[l].Kpad;
        pack_weights<<<ceil_div(total, 256), 256, 0, stream>>>(
            (const float*)d_in[1 + 2 * l], wp[l].pk,
            wp[l].Cout, wp[l].Cin, wp[l].Kreal, wp[l].Kpad, wp[l].CoutPad);
    }

    auto conv = [&](const _Float16* xin, int layer, _Float16* o16, float* o32,
                    int H, int W, int relu) {
        const WP& L = wp[layer];
        int tiles = B * (L.CoutPad >> 4) * ((H * W) / (16 * NT));
        int blocks = ceil_div(tiles, 8);
        const float* bias = (const float*)d_in[2 + 2 * layer];
        if (L.KH == 1) {
            conv1x1_wmma<NT, 256><<<blocks, 256, 0, stream>>>(
                xin, L.pk, bias, o16, o32, B, L.Cout, L.CoutPad, H, W,
                relu, o32 != nullptr);
        } else if (L.Cin == 1) {
            conv3x3_c1_wmma<NT><<<blocks, 256, 0, stream>>>(
                xin, L.pk, bias, o16, L.Cout, H, W, B);
        } else if (L.Cin == 64) {
            conv3x3_wmma<NT, 6><<<blocks, 256, 0, stream>>>(
                xin, L.pk, bias, o16, o32, zerobuf, B, L.Cout, L.CoutPad, H, W,
                relu, o32 != nullptr);
        } else {
            conv3x3_wmma<NT, 7><<<blocks, 256, 0, stream>>>(
                xin, L.pk, bias, o16, o32, zerobuf, B, L.Cout, L.CoutPad, H, W,
                relu, o32 != nullptr);
        }
    };

    // ---- backbone (all activations NHWC f16) ----
    conv(img16, 0, act0, nullptr, 480, 640, 1);                 // conv1a
    conv(act0,  1, act1, nullptr, 480, 640, 1);                 // conv1b
    pool2_f16<<<ceil_div(B*64*240*320, 256), 256, 0, stream>>>(act1, act0, B, 64, 480, 640);
    conv(act0,  2, act1, nullptr, 240, 320, 1);                 // conv2a
    conv(act1,  3, act0, nullptr, 240, 320, 1);                 // conv2b
    pool2_f16<<<ceil_div(B*64*120*160, 256), 256, 0, stream>>>(act0, act1, B, 64, 240, 320);
    conv(act1,  4, act0, nullptr, 120, 160, 1);                 // conv3a (64->128)
    conv(act0,  5, act1, nullptr, 120, 160, 1);                 // conv3b
    pool2_f16<<<ceil_div(B*128*60*80, 256), 256, 0, stream>>>(act1, act0, B, 128, 120, 160);
    conv(act0,  6, act1, nullptr, 60, 80, 1);                   // conv4a
    conv(act1,  7, act0, nullptr, 60, 80, 1);                   // conv4b -> trunk in act0

    // ---- detector head ----
    conv(act0,  8, act1, nullptr, 60, 80, 1);                   // convPa
    conv(act1,  9, nullptr, scraw, 60, 80, 0);                  // convPb -> [B,hw,65]
    score_head<<<ceil_div(B*60*80, 128), 128, 0, stream>>>(scraw, out_sc, B, 60, 80);

    // ---- NMS ----
    int nb = ceil_div(NPIX, 256);
    maxpool7<<<nb, 256, 0, stream>>>(out_sc, T0, B, H0, W0);
    nms_mask_init<<<nb, 256, 0, stream>>>(out_sc, T0, Mb, NPIX);
    for (int r = 0; r < 2; ++r) {
        maxpool7<<<nb, 256, 0, stream>>>(Mb, T0, B, H0, W0);
        nms_supp<<<nb, 256, 0, stream>>>(out_sc, T0, S1, NPIX);
        maxpool7<<<nb, 256, 0, stream>>>(S1, T1, B, H0, W0);
        nms_update<<<nb, 256, 0, stream>>>(Mb, S1, T1, T0, NPIX);
    }
    nms_final<<<nb, 256, 0, stream>>>(out_sc, Mb, out_nms, NPIX);

    // ---- descriptor head ----
    conv(act0, 10, act1, nullptr, 60, 80, 1);                   // convDa
    conv(act1, 11, nullptr, descRaw, 60, 80, 0);                // convDb -> [B,hw,256]
    desc_norm_t<<<ceil_div(B*60*80, 128), 128, 0, stream>>>(descRaw, out_desc, B, 256, 60*80);
}
